// RNNModel_65893388255910
// MI455X (gfx1250) — matrix operand order repaired and checked by reading, compiled, not run on hardware
//
#include <hip/hip_runtime.h>

// MI455X / gfx1250, wave32. BiGRU layer-0 scan done with chained
// v_wmma_f32_16x16x32_f16 (K=32 == H), one wave per 16-row batch tile,
// one wave per direction, full 512-step recurrence on-chip.

typedef __attribute__((ext_vector_type(16))) _Float16 v16h;
typedef __attribute__((ext_vector_type(8)))  float    v8f;

#define HDIM   32
#define EMBD   10
#define TLEN   512
#define BATCH  1024
#define VOCAB  5
#define HSTRIDE 40   // padded halfs per row in LDS staging (16B aligned, conflict-free)

__device__ __forceinline__ float fast_rcp(float x) {
#if __has_builtin(__builtin_amdgcn_rcpf)
    return __builtin_amdgcn_rcpf(x);      // single v_rcp_f32 (TRANS, co-executes)
#else
    return 1.0f / x;
#endif
}
__device__ __forceinline__ float sigf(float x) {
    return fast_rcp(1.0f + __expf(-x));   // v_exp_f32 + v_rcp_f32
}
__device__ __forceinline__ float tanh_fast(float x) {
#if __has_builtin(__builtin_amdgcn_tanhf)
    return __builtin_amdgcn_tanhf(x);     // gfx1250 hardware v_tanh_f32
#else
    x = fminf(15.0f, fmaxf(-15.0f, x));
    float e = __expf(2.0f * x);
    return (e - 1.0f) * fast_rcp(e + 1.0f);
#endif
}

// ---------------------------------------------------------------------------
// x_embed = emb[ids]  (pure gather, bandwidth bound: ~21MB write)
// ---------------------------------------------------------------------------
__global__ void embed_kernel(const float* __restrict__ emb,
                             const int*   __restrict__ ids,
                             float*       __restrict__ out) {
    int i = blockIdx.x * blockDim.x + threadIdx.x;
    const int N = BATCH * TLEN * EMBD;
    if (i >= N) return;
    int e  = i % EMBD;
    int bt = i / EMBD;
    out[i] = emb[ids[bt] * EMBD + e];
}

// ---------------------------------------------------------------------------
// Layer-0 BiGRU scan. grid = B/16 blocks of 64 threads (2 waves).
// wave 0: forward direction, wave 1: backward direction, same 16 batch rows.
// ---------------------------------------------------------------------------
__global__ __launch_bounds__(64) void gru_l0_kernel(
    const float* __restrict__ Wi_f, const float* __restrict__ Wh_f,
    const float* __restrict__ bi_f, const float* __restrict__ bh_f,
    const float* __restrict__ Wi_b, const float* __restrict__ Wh_b,
    const float* __restrict__ bi_b, const float* __restrict__ bh_b,
    const float* __restrict__ emb,
    const int*   __restrict__ ids,   // [B,T]
    const int*   __restrict__ lens,  // [B]
    float*       __restrict__ hid_out) // [B,64] = [hf | hb]
{
    __shared__ int      ids_s[16 * 513];           // padded: bank-conflict-free
    __shared__ _Float16 hbuf[2][16 * HSTRIDE];     // per-wave h staging

    const int b0   = blockIdx.x * 16;
    const int tid  = threadIdx.x;
    const int wave = tid >> 5;                     // 0 = fwd, 1 = bwd
    const int lane = tid & 31;

    // ---- preload token ids for our 16 rows, all timesteps (coalesced) ----
    for (int i = tid; i < 16 * TLEN; i += 64) {
        int m = i >> 9;            // i / TLEN
        int t = i & (TLEN - 1);
        ids_s[m * 513 + t] = ids[(b0 + m) * TLEN + t];
    }
    __syncthreads();

    const float* Wi = wave ? Wi_b : Wi_f;
    const float* Wh = wave ? Wh_b : Wh_f;
    const float* bi = wave ? bi_b : bi_f;
    const float* bh = wave ? bh_b : bh_f;

    // ---- build B fragments in registers -------------------------------
    // B-frag K order chosen here (contiguous, same as A; 16-bit, 32x16): lane L holds column n = L%16;
    //   lanes 0-15: halfs h -> K = h (0..15); lanes 16-31: halfs h -> K = 16+h.
    const int n   = lane & 15;
    const int khi = (lane >> 4) * 16;

    v16h b1[6];   // Wh^T tiles         (B1[k][g] = Wh[g][k])
    v16h b2[6];   // XWtok tiles        (rows 0..4 = emb[v] @ Wi^T, rest 0)
    v8f  c_rz[4]; // C init for chained r/z tiles:  splat(bh[g] + bi[g])
    v8f  c_bh[2]; // C init for gh_n tiles:         splat(bh[g])
    v8f  c_bi[2]; // C init for xw_n tiles:         splat(bi[g])

    #pragma unroll
    for (int j = 0; j < 6; ++j) {
        const int g = 16 * j + n;                  // output column 0..95
        #pragma unroll
        for (int h = 0; h < 16; ++h)
            b1[j][h] = (_Float16)Wh[g * HDIM + khi + h];

        #pragma unroll
        for (int h = 0; h < 16; ++h) b2[j][h] = (_Float16)0.0f;
        if (lane < 16) {                           // K rows 0..4 live here
            #pragma unroll
            for (int v = 0; v < VOCAB; ++v) {
                float s = 0.0f;
                #pragma unroll
                for (int e = 0; e < EMBD; ++e)
                    s += emb[v * EMBD + e] * Wi[g * EMBD + e];
                b2[j][v] = (_Float16)s;
            }
        }
        const float bhg = bh[g], big = bi[g];
        if (j < 4) {
            #pragma unroll
            for (int r = 0; r < 8; ++r) c_rz[j][r] = bhg + big;
        } else {
            #pragma unroll
            for (int r = 0; r < 8; ++r) { c_bh[j - 4][r] = bhg; c_bi[j - 4][r] = big; }
        }
    }

    // ---- elementwise-layout state (D-tile layout) ----------------------
    // lane L handles rows m = 8*(L/16)+i (i=0..7), cols k = L%16 and L%16+16
    const int rhalf = lane >> 4;
    const int col   = lane & 15;
    int lenv[8];
    #pragma unroll
    for (int i = 0; i < 8; ++i) lenv[i] = lens[b0 + rhalf * 8 + i];

    float hA[8], hB[8];
    #pragma unroll
    for (int i = 0; i < 8; ++i) { hA[i] = 0.0f; hB[i] = 0.0f; }

    v16h a_h;
    #pragma unroll
    for (int i = 0; i < 16; ++i) a_h[i] = (_Float16)0.0f;

    _Float16* hs = &hbuf[wave][0];
    const int arow = lane & 15;                    // A-frag: row = L%16
    const int alo  = (lane >> 4) * 16;             // A-frag K sub-offset

    for (int step = 0; step < TLEN; ++step) {
        const int t = wave ? (TLEN - 1 - step) : step;

        // one-hot A2 fragment (token of row arow; K=0..4 live in lanes 0-15)
        v16h a2;
        #pragma unroll
        for (int i = 0; i < 16; ++i) a2[i] = (_Float16)0.0f;
        const int tok = ids_s[arow * 513 + t];
        if (lane < 16) {
            #pragma unroll
            for (int v = 0; v < VOCAB; ++v)
                a2[v] = (tok == v) ? (_Float16)1.0f : (_Float16)0.0f;
        }

        // r/z pregates: chained WMMA  P = onehot*XWtok + (h*Wh^T + (bh+bi))
        v8f P[4];
        #pragma unroll
        for (int j = 0; j < 4; ++j) {
            v8f g = __builtin_amdgcn_wmma_f32_16x16x32_f16(
                        false, a_h, false, b1[j], (short)0, c_rz[j], false, false);
            P[j]  = __builtin_amdgcn_wmma_f32_16x16x32_f16(
                        false, a2,  false, b2[j], (short)0, g,       false, false);
        }
        // n gate needs xw_n and gh_n separately (r scales only gh_n)
        v8f G4 = __builtin_amdgcn_wmma_f32_16x16x32_f16(
                    false, a_h, false, b1[4], (short)0, c_bh[0], false, false);
        v8f G5 = __builtin_amdgcn_wmma_f32_16x16x32_f16(
                    false, a_h, false, b1[5], (short)0, c_bh[1], false, false);
        v8f X4 = __builtin_amdgcn_wmma_f32_16x16x32_f16(
                    false, a2,  false, b2[4], (short)0, c_bi[0], false, false);
        v8f X5 = __builtin_amdgcn_wmma_f32_16x16x32_f16(
                    false, a2,  false, b2[5], (short)0, c_bi[1], false, false);

        // gates + packed-sequence masking (rcp/tanh TRANS ops, no IEEE div)
        #pragma unroll
        for (int i = 0; i < 8; ++i) {
            const bool msk = t < lenv[i];
            float rA  = sigf(P[0][i]);
            float zA  = sigf(P[2][i]);
            float nA  = tanh_fast(X4[i] + rA * G4[i]);
            float hnA = (1.0f - zA) * nA + zA * hA[i];
            hA[i] = msk ? hnA : hA[i];

            float rB  = sigf(P[1][i]);
            float zB  = sigf(P[3][i]);
            float nB  = tanh_fast(X5[i] + rB * G5[i]);
            float hnB = (1.0f - zB) * nB + zB * hB[i];
            hB[i] = msk ? hnB : hB[i];
        }

        // D-layout f32  ->  A-fragment f16 via LDS staging (intra-wave)
        #pragma unroll
        for (int i = 0; i < 8; ++i) {
            const int m = rhalf * 8 + i;
            hs[m * HSTRIDE + col]      = (_Float16)hA[i];
            hs[m * HSTRIDE + col + 16] = (_Float16)hB[i];
        }
        // LDS is in-order within a wave; fence stops compiler reordering too
        asm volatile("s_wait_dscnt 0" ::: "memory");
        const _Float16* arp = hs + arow * HSTRIDE + alo;
        #pragma unroll
        for (int i = 0; i < 8; ++i) {
            a_h[i]     = arp[i];        // K = alo + i
            a_h[8 + i] = arp[8 + i];    // K = alo + 8 + i
        }
    }

    // final hidden state  ->  hid_out[b][dir*32 + k]
    #pragma unroll
    for (int i = 0; i < 8; ++i) {
        const int b = b0 + rhalf * 8 + i;
        hid_out[b * 64 + wave * HDIM + col]      = hA[i];
        hid_out[b * 64 + wave * HDIM + col + 16] = hB[i];
    }
}

// ---------------------------------------------------------------------------
// FC head: logits = (hid @ fc1^T + b1) @ fc2^T + b2   (no nonlinearity)
// ---------------------------------------------------------------------------
__global__ void head_kernel(const float* __restrict__ hid,
                            const float* __restrict__ fc1w, const float* __restrict__ fc1b,
                            const float* __restrict__ fc2w, const float* __restrict__ fc2b,
                            float* __restrict__ logits) {
    int b = blockIdx.x * blockDim.x + threadIdx.x;
    if (b >= BATCH) return;
    const float* hv = hid + b * 64;
    float h1[32];
    #pragma unroll 4
    for (int o = 0; o < 32; ++o) {
        float s = fc1b[o];
        #pragma unroll
        for (int k = 0; k < 64; ++k) s += fc1w[o * 64 + k] * hv[k];
        h1[o] = s;
    }
    #pragma unroll
    for (int c = 0; c < 2; ++c) {
        float s = fc2b[c];
        #pragma unroll
        for (int o = 0; o < 32; ++o) s += fc2w[c * 32 + o] * h1[o];
        logits[b * 2 + c] = s;
    }
}

// ---------------------------------------------------------------------------
extern "C" void kernel_launch(void* const* d_in, const int* in_sizes, int n_in,
                              void* d_out, int out_size, void* d_ws, size_t ws_size,
                              hipStream_t stream) {
    // setup_inputs() flat order:
    // 0:emb 1:Wi_l0_f 2:Wh_l0_f 3:bi_l0_f 4:bh_l0_f 5:Wi_l0_b 6:Wh_l0_b
    // 7:bi_l0_b 8:bh_l0_b 9..16: layer-1 (dead) 17:fc1_w 18:fc1_b 19:fc2_w
    // 20:fc2_b 21:batch_data 22:batch_lens 23:batch_size
    const float* emb  = (const float*)d_in[0];
    const float* Wi_f = (const float*)d_in[1];
    const float* Wh_f = (const float*)d_in[2];
    const float* bi_f = (const float*)d_in[3];
    const float* bh_f = (const float*)d_in[4];
    const float* Wi_b = (const float*)d_in[5];
    const float* Wh_b = (const float*)d_in[6];
    const float* bi_b = (const float*)d_in[7];
    const float* bh_b = (const float*)d_in[8];
    const float* fc1w = (const float*)d_in[17];
    const float* fc1b = (const float*)d_in[18];
    const float* fc2w = (const float*)d_in[19];
    const float* fc2b = (const float*)d_in[20];
    const int*   ids  = (const int*)d_in[21];
    const int*   lens = (const int*)d_in[22];

    float* logits  = (float*)d_out;                 // [1,B,2]  = 2048 floats
    float* x_embed = logits + BATCH * 2;            // [B,T,EMB]
    float* hid     = (float*)d_ws;                  // [B,64] scratch

    const int Nemb = BATCH * TLEN * EMBD;
    embed_kernel<<<(Nemb + 255) / 256, 256, 0, stream>>>(emb, ids, x_embed);

    gru_l0_kernel<<<BATCH / 16, 64, 0, stream>>>(
        Wi_f, Wh_f, bi_f, bh_f, Wi_b, Wh_b, bi_b, bh_b,
        emb, ids, lens, hid);

    head_kernel<<<(BATCH + 255) / 256, 256, 0, stream>>>(
        hid, fc1w, fc1b, fc2w, fc2b, logits);
}
